// ComplexMultiheadAttention_73229192396898
// MI455X (gfx1250) — matrix operand order repaired and checked by reading, not compiled, not run on hardware
//
#include <hip/hip_runtime.h>

typedef __bf16 bf16;
typedef __attribute__((ext_vector_type(16))) __bf16 v16bf;
typedef __attribute__((ext_vector_type(8)))  __bf16 v8bf;
typedef __attribute__((ext_vector_type(8)))  float  v8f;

typedef __attribute__((ext_vector_type(4))) unsigned int tdm_g0_t;  // D# group0
typedef __attribute__((ext_vector_type(8))) int          tdm_g1_t;  // D# group1
typedef __attribute__((ext_vector_type(4))) int          tdm_g2_t;  // D# group2/3
typedef __attribute__((ext_vector_type(8))) int          tdm_g4_t;  // extra arg (clang-23 form)

#define B_SZ   2
#define L_SEQ  4096
#define EMB    512
#define HEADS  8
#define DH     64
#define WIN    128
#define MROWS  (B_SZ * L_SEQ)   // 8192

union V16U { v16bf v; v8bf h[2]; };
union BFU  { bf16 b; unsigned short s; };

__device__ inline bf16 f2bf(float f) { return (bf16)f; }
__device__ inline unsigned short bfbits(bf16 b) { BFU u; u.b = b; return u.s; }

// A-matrix 16x32 bf16 fragment (wave32): lane m = l&15; half = l>>4
__device__ inline v16bf load_a_frag(const bf16* base, int stride, int lane) {
  const int m = lane & 15, hl = lane >> 4;
  V16U u;
  u.h[0] = *(const v8bf*)(base + (size_t)m * stride + hl * 8);
  u.h[1] = *(const v8bf*)(base + (size_t)m * stride + 16 + hl * 8);
  return u.v;
}
// B-matrix 32x16 bf16 fragment: lane n = l&15 holds one column; k = half*8 + e (e<8), 16 + half*8 + (e-8) (e>=8)
__device__ inline v16bf load_b_frag(const bf16* base, int stride, int lane) {
  const int n = lane & 15, hl = lane >> 4;
  V16U u;
  u.h[0] = *(const v8bf*)(base + (size_t)n * stride + hl * 8);
  u.h[1] = *(const v8bf*)(base + (size_t)n * stride + 16 + hl * 8);
  return u.v;
}

__device__ inline v8f wmma_bf16(v16bf a, v16bf b, v8f c) {
  return __builtin_amdgcn_wmma_f32_16x16x32_bf16(false, a, false, b, (short)0, c, false, false);
}

// ---------------------------------------------------------------------------
// TDM: async-load one 128x32 bf16 tile (row stride 512 elems in global)
// into LDS with hardware padding: pad_interval = 16 DWORDs (64B row),
// pad_amount = 4 DWORDs (16B) -> LDS row stride 80B == (32+8) bf16.
// 6-arg builtin form (clang-23 / therock-10.0 headers).
// ---------------------------------------------------------------------------
__device__ inline void tdm_load_tile(unsigned lds_off, const bf16* gptr, int dim1) {
  const unsigned long long ga = (unsigned long long)(size_t)gptr;
  tdm_g0_t g0;
  g0[0] = 1u;                                         // count=1, user mode
  g0[1] = lds_off;                                    // lds_addr (bytes)
  g0[2] = (unsigned)(ga & 0xffffffffu);               // global_addr[31:0]
  g0[3] = (unsigned)((ga >> 32) & 0x01ffffffu)        // global_addr[56:32]
          | 0x80000000u;                              // type=2 ("image")
  tdm_g1_t g1;
  g1[0] = (1 << 16)      // data_size = 2 bytes
        | (1 << 20)      // pad_enable
        | (3 << 22)      // pad_interval: 16 DWORDs
        | (3 << 25);     // pad_amount: 4 DWORDs
  g1[1] = (int)(512u << 16);                          // tensor_dim0 = 512 (bits 79:48)
  g1[2] = (int)(((unsigned)dim1 & 0xffffu) << 16);    // tensor_dim1 lo16 (bits 111:80)
  g1[3] = (int)((((unsigned)dim1 >> 16) & 0xffffu)    // tensor_dim1 hi16
        | (32u << 16));                               // tile_dim0 = 32 (bits 127:112)
  g1[4] = 128;                                        // tile_dim1 = 128 (bits 143:128)
  g1[5] = 512;                                        // tensor_dim0_stride (bits 207:160)
  g1[6] = 0;
  g1[7] = 0;
  const tdm_g2_t gz4 = {0, 0, 0, 0};
  const tdm_g4_t gz8 = {0, 0, 0, 0, 0, 0, 0, 0};
  __builtin_amdgcn_tensor_load_to_lds(g0, g1, gz4, gz4, gz8, 0);
}

// ---------------------------------------------------------------------------
// f32 -> bf16 conversion (one-time pre-pass for activations & weights)
// ---------------------------------------------------------------------------
__global__ __launch_bounds__(256) void cvt_kernel(const float* __restrict__ in,
                                                  bf16* __restrict__ out, int n4) {
  const int i = blockIdx.x * 256 + threadIdx.x;
  if (i < n4) {
    const float4 v = *(const float4*)(in + (size_t)i * 4);
    const unsigned a = bfbits(f2bf(v.x)), b = bfbits(f2bf(v.y));
    const unsigned c = bfbits(f2bf(v.z)), d = bfbits(f2bf(v.w));
    uint2 pk;
    pk.x = a | (b << 16);
    pk.y = c | (d << 16);
    *(uint2*)((unsigned short*)out + (size_t)i * 4) = pk;
  }
}

// ---------------------------------------------------------------------------
// Pure-bf16 GEMM: Out[M,N] = X[M,512] @ W[N,512]^T + bias, M=8192, N=512.
// TDM double-buffered LDS staging (wave 0 drives the DMA engine; TENSORcnt
// is per-wave so only wave 0 waits, barrier publishes to the workgroup).
// OM: 0 = bf16 row-major, 1 = bf16 head-transposed [B][H][D][L], 2 = f32.
// ---------------------------------------------------------------------------
template<int OM>
__global__ __launch_bounds__(256) void gemm_bf16(const bf16* __restrict__ X,
                                                 const bf16* __restrict__ W,
                                                 const float* __restrict__ bias,
                                                 void* __restrict__ Outv) {
  constexpr int KD = EMB, ND = EMB, BK = 32, LDX = BK + 8;  // 80B LDS row
  __shared__ __attribute__((aligned(16))) bf16 sX[2][128 * LDX];
  __shared__ __attribute__((aligned(16))) bf16 sW[2][128 * LDX];

  const int tid = threadIdx.x;
  const int lane = tid & 31, wave = tid >> 5;
  const int waveM = wave >> 2, waveN = wave & 3;
  const int m0 = blockIdx.y * 128, n0 = blockIdx.x * 128;
  const int n = lane & 15, hl = lane >> 4;

  const v8f vzero = {0.f, 0.f, 0.f, 0.f, 0.f, 0.f, 0.f, 0.f};
  v8f acc[4][2];
  #pragma unroll
  for (int i = 0; i < 4; ++i)
    #pragma unroll
    for (int j = 0; j < 2; ++j) acc[i][j] = vzero;

  // LDS aperture maps flat addr[31:0] to the LDS byte offset (ISA 10.2)
  const unsigned ldsX[2] = {(unsigned)(size_t)(void*)&sX[0][0],
                            (unsigned)(size_t)(void*)&sX[1][0]};
  const unsigned ldsW[2] = {(unsigned)(size_t)(void*)&sW[0][0],
                            (unsigned)(size_t)(void*)&sW[1][0]};

  if (wave == 0) {
    tdm_load_tile(ldsX[0], X + (size_t)m0 * KD, MROWS);
    tdm_load_tile(ldsW[0], W + (size_t)n0 * KD, ND);
    __builtin_amdgcn_s_wait_tensorcnt(0);
  }
  __syncthreads();

  constexpr int NIT = KD / BK;  // 16
  for (int it = 0; it < NIT; ++it) {
    const int cur = it & 1;
    if (wave == 0 && (it + 1) < NIT) {       // prefetch next K-panel via TDM
      const int kk = (it + 1) * BK;
      tdm_load_tile(ldsX[cur ^ 1], X + (size_t)m0 * KD + kk, MROWS);
      tdm_load_tile(ldsW[cur ^ 1], W + (size_t)n0 * KD + kk, ND);
    }

    v16bf aF[4], bF[2];
    #pragma unroll
    for (int ms = 0; ms < 4; ++ms)
      aF[ms] = load_a_frag(&sX[cur][(waveM * 64 + ms * 16) * LDX], LDX, lane);
    #pragma unroll
    for (int ns = 0; ns < 2; ++ns)
      bF[ns] = load_b_frag(&sW[cur][(waveN * 32 + ns * 16) * LDX], LDX, lane);
    #pragma unroll
    for (int ms = 0; ms < 4; ++ms)
      #pragma unroll
      for (int ns = 0; ns < 2; ++ns)
        acc[ms][ns] = wmma_bf16(aF[ms], bF[ns], acc[ms][ns]);

    if (wave == 0 && (it + 1) < NIT) __builtin_amdgcn_s_wait_tensorcnt(0);
    __syncthreads();
  }

  #pragma unroll
  for (int ms = 0; ms < 4; ++ms) {
    #pragma unroll
    for (int ns = 0; ns < 2; ++ns) {
      const int ncol  = n0 + waveN * 32 + ns * 16 + n;
      const float bv  = bias[ncol];
      const int mrow0 = m0 + waveM * 64 + ms * 16 + hl * 8;  // 8 consecutive M
      if constexpr (OM == 0) {
        bf16* out = (bf16*)Outv;
        #pragma unroll
        for (int r = 0; r < 8; ++r)
          out[(size_t)(mrow0 + r) * ND + ncol] = f2bf(acc[ms][ns][r] + bv);
      } else if constexpr (OM == 1) {
        // head-transposed store for V: [B][H][D][L]; lane's 8 M are contiguous
        unsigned short* out = (unsigned short*)Outv;
        const int hh = ncol >> 6, dd = ncol & 63;
        const int bb = mrow0 >> 12, ls = mrow0 & 4095;
        const size_t addr = (((size_t)bb * HEADS + hh) * DH + dd) * L_SEQ + ls;
        unsigned short tmp[8];
        #pragma unroll
        for (int r = 0; r < 8; ++r) tmp[r] = bfbits(f2bf(acc[ms][ns][r] + bv));
        int4 pk;
        pk.x = tmp[0] | (tmp[1] << 16);
        pk.y = tmp[2] | (tmp[3] << 16);
        pk.z = tmp[4] | (tmp[5] << 16);
        pk.w = tmp[6] | (tmp[7] << 16);
        *(int4*)(out + addr) = pk;
      } else {
        float* out = (float*)Outv;
        #pragma unroll
        for (int r = 0; r < 8; ++r)
          out[(size_t)(mrow0 + r) * ND + ncol] = acc[ms][ns][r] + bv;
      }
    }
  }
}

// ---------------------------------------------------------------------------
// Flash attention with band mask (mask only j <= i-W-1; NO causal upper bound).
// One wave owns 16 query rows; 8 independent waves per block; 64-key tiles.
// scores = (qr@kr^T + qi@ki^T)/sqrt(D) ; online softmax ; O += P@V.
// ---------------------------------------------------------------------------
__global__ __launch_bounds__(256) void attn_kernel(const bf16* __restrict__ qr,
                                                   const bf16* __restrict__ qi,
                                                   const bf16* __restrict__ kr,
                                                   const bf16* __restrict__ ki,
                                                   const bf16* __restrict__ vt,
                                                   bf16* __restrict__ o) {
  __shared__ __attribute__((aligned(16))) bf16 sP[8 * 16 * 64];  // per-wave P tile
  const int tid = threadIdx.x;
  const int lane = tid & 31, wave = tid >> 5;
  const int n = lane & 15, hl = lane >> 4;
  const int bh = blockIdx.y, b = bh >> 3, h = bh & 7;
  const int i0 = blockIdx.x * 128 + wave * 16;

  const size_t qbase = ((size_t)(b * L_SEQ + i0)) * EMB + h * DH;
  v16bf qrF[2], qiF[2];
  qrF[0] = load_a_frag(qr + qbase,      EMB, lane);
  qrF[1] = load_a_frag(qr + qbase + 32, EMB, lane);
  qiF[0] = load_a_frag(qi + qbase,      EMB, lane);
  qiF[1] = load_a_frag(qi + qbase + 32, EMB, lane);

  const v8f vzero = {0.f, 0.f, 0.f, 0.f, 0.f, 0.f, 0.f, 0.f};
  v8f Oa[4];
  #pragma unroll
  for (int t = 0; t < 4; ++t) Oa[t] = vzero;
  float mrow[8], lrow[8];
  #pragma unroll
  for (int r = 0; r < 8; ++r) { mrow[r] = -1e30f; lrow[r] = 0.f; }

  int jt = i0 - WIN; if (jt < 0) jt = 0; jt &= ~63;   // skip fully-masked tiles
  const float scale = 0.125f;                          // 1/sqrt(64)
  const bf16* vb = vt + (size_t)bh * DH * L_SEQ;
  bf16* pbuf = &sP[wave * 16 * 64];

  for (int j0 = jt; j0 < L_SEQ; j0 += 64) {
    v8f S[4];
    #pragma unroll
    for (int t = 0; t < 4; ++t) S[t] = vzero;
    #pragma unroll
    for (int t = 0; t < 4; ++t) {
      const size_t kb = ((size_t)(b * L_SEQ + j0 + t * 16)) * EMB + h * DH;
      #pragma unroll
      for (int s = 0; s < 2; ++s) {
        v16bf bKr = load_b_frag(kr + kb + s * 32, EMB, lane);
        S[t] = wmma_bf16(qrF[s], bKr, S[t]);
        v16bf bKi = load_b_frag(ki + kb + s * 32, EMB, lane);
        S[t] = wmma_bf16(qiF[s], bKi, S[t]);
      }
    }
    // scale + band mask (masked iff i - j >= W+1)
    #pragma unroll
    for (int t = 0; t < 4; ++t) {
      const int j = j0 + t * 16 + n;
      #pragma unroll
      for (int r = 0; r < 8; ++r) {
        float s = S[t][r] * scale;
        const int i = i0 + hl * 8 + r;
        if (i - j > WIN) s = -1e30f;
        S[t][r] = s;
      }
    }
    // online softmax (rows live in 16-lane groups; xor<16 stays in-group)
    #pragma unroll
    for (int r = 0; r < 8; ++r) {
      float mx = fmaxf(fmaxf(S[0][r], S[1][r]), fmaxf(S[2][r], S[3][r]));
      #pragma unroll
      for (int d = 1; d < 16; d <<= 1) mx = fmaxf(mx, __shfl_xor(mx, d, 32));
      const float mnew = fmaxf(mrow[r], mx);
      const float corr = __expf(mrow[r] - mnew);
      mrow[r] = mnew;
      float rs = 0.f;
      #pragma unroll
      for (int t = 0; t < 4; ++t) {
        const float p = __expf(S[t][r] - mnew);
        S[t][r] = p;
        rs += p;
      }
      #pragma unroll
      for (int d = 1; d < 16; d <<= 1) rs += __shfl_xor(rs, d, 32);
      lrow[r] = lrow[r] * corr + rs;
      #pragma unroll
      for (int t = 0; t < 4; ++t) Oa[t][r] *= corr;
    }
    // P: C-layout -> LDS row-major [m][k]
    #pragma unroll
    for (int t = 0; t < 4; ++t)
      #pragma unroll
      for (int r = 0; r < 8; ++r)
        pbuf[(hl * 8 + r) * 64 + t * 16 + n] = f2bf(S[t][r]);
    // O += P @ V   (V is head-transposed: B-frags are contiguous b128 loads)
    #pragma unroll
    for (int s = 0; s < 2; ++s) {
      v16bf aP = load_a_frag(pbuf + s * 32, 64, lane);
      #pragma unroll
      for (int t = 0; t < 4; ++t) {
        v16bf bV = load_b_frag(vb + (size_t)(t * 16) * L_SEQ + j0 + s * 32, L_SEQ, lane);
        Oa[t] = wmma_bf16(aP, bV, Oa[t]);
      }
    }
  }

  #pragma unroll
  for (int r = 0; r < 8; ++r) {
    const float inv = 1.f / lrow[r];
    const size_t orow = ((size_t)(b * L_SEQ + i0 + hl * 8 + r)) * EMB + h * DH;
    #pragma unroll
    for (int t = 0; t < 4; ++t)
      o[orow + t * 16 + n] = f2bf(Oa[t][r] * inv);
  }
}

// ---------------------------------------------------------------------------
extern "C" void kernel_launch(void* const* d_in, const int* in_sizes, int n_in,
                              void* d_out, int out_size, void* d_ws, size_t ws_size,
                              hipStream_t stream) {
  const float* query = (const float*)d_in[0];
  const float* keyx  = (const float*)d_in[1];
  const float* value = (const float*)d_in[2];
  const float* qWr = (const float*)d_in[3],  *qWi = (const float*)d_in[4];
  const float* qbr = (const float*)d_in[5],  *qbi = (const float*)d_in[6];
  const float* kWr = (const float*)d_in[7],  *kWi = (const float*)d_in[8];
  const float* kbr = (const float*)d_in[9],  *kbi = (const float*)d_in[10];
  const float* vWr = (const float*)d_in[11], *vWi = (const float*)d_in[12];
  const float* vbr = (const float*)d_in[13], *vbi = (const float*)d_in[14];
  const float* oWr = (const float*)d_in[15], *oWi = (const float*)d_in[16];
  const float* obr = (const float*)d_in[17], *obi = (const float*)d_in[18];
  (void)vWi; (void)vbi; (void)in_sizes; (void)n_in; (void)out_size; (void)ws_size;

  char* p = (char*)d_ws;
  const size_t ACT = (size_t)MROWS * EMB * sizeof(bf16);  // 8 MB
  const size_t WSZ = (size_t)EMB * EMB * sizeof(bf16);    // 512 KB
  bf16* qbuf = (bf16*)p; p += ACT;
  bf16* kbuf = (bf16*)p; p += ACT;
  bf16* vbuf = (bf16*)p; p += ACT;
  bf16* wqr = (bf16*)p; p += WSZ;
  bf16* wqi = (bf16*)p; p += WSZ;
  bf16* wkr = (bf16*)p; p += WSZ;
  bf16* wki = (bf16*)p; p += WSZ;
  bf16* wvr = (bf16*)p; p += WSZ;
  bf16* wor = (bf16*)p; p += WSZ;
  bf16* woi = (bf16*)p; p += WSZ;
  bf16* qrb = (bf16*)p; p += ACT;
  bf16* qib = (bf16*)p; p += ACT;
  bf16* krb = (bf16*)p; p += ACT;
  bf16* kib = (bf16*)p; p += ACT;
  bf16* vtb = (bf16*)p; p += ACT;   // [B][H][D][L]
  bf16* ob  = (bf16*)p; p += ACT;

  dim3 blk(256);
  // one-time f32 -> bf16 pre-pass (activations + the 7 live weight matrices)
  const int nA4 = (int)((size_t)MROWS * EMB / 4), gA = nA4 / 256;
  const int nW4 = (int)((size_t)EMB * EMB / 4),   gW = nW4 / 256;
  cvt_kernel<<<gA, blk, 0, stream>>>(query, qbuf, nA4);
  cvt_kernel<<<gA, blk, 0, stream>>>(keyx,  kbuf, nA4);
  cvt_kernel<<<gA, blk, 0, stream>>>(value, vbuf, nA4);
  cvt_kernel<<<gW, blk, 0, stream>>>(qWr, wqr, nW4);
  cvt_kernel<<<gW, blk, 0, stream>>>(qWi, wqi, nW4);
  cvt_kernel<<<gW, blk, 0, stream>>>(kWr, wkr, nW4);
  cvt_kernel<<<gW, blk, 0, stream>>>(kWi, wki, nW4);
  cvt_kernel<<<gW, blk, 0, stream>>>(vWr, wvr, nW4);
  cvt_kernel<<<gW, blk, 0, stream>>>(oWr, wor, nW4);
  cvt_kernel<<<gW, blk, 0, stream>>>(oWi, woi, nW4);

  dim3 gg(EMB / 128, MROWS / 128);
  gemm_bf16<0><<<gg, blk, 0, stream>>>(qbuf, wqr, qbr, qrb);
  gemm_bf16<0><<<gg, blk, 0, stream>>>(qbuf, wqi, qbi, qib);
  gemm_bf16<0><<<gg, blk, 0, stream>>>(kbuf, wkr, kbr, krb);
  gemm_bf16<0><<<gg, blk, 0, stream>>>(kbuf, wki, kbi, kib);
  gemm_bf16<1><<<gg, blk, 0, stream>>>(vbuf, wvr, vbr, vtb);

  attn_kernel<<<dim3(L_SEQ / 128, B_SZ * HEADS), blk, 0, stream>>>(qrb, qib, krb, kib, vtb, ob);

  float* outr = (float*)d_out;
  float* outi = outr + (size_t)MROWS * EMB;
  gemm_bf16<2><<<gg, blk, 0, stream>>>(ob, wor, obr, outr);
  gemm_bf16<2><<<gg, blk, 0, stream>>>(ob, woi, obi, outi);
}
